// FeatureDistillation_module_37726992728548
// MI455X (gfx1250) — matrix operand order repaired and checked by reading, not compiled, not run on hardware
//
#include <hip/hip_runtime.h>

typedef __bf16 bf16;
typedef __attribute__((ext_vector_type(16))) __bf16 v16bf;
typedef __attribute__((ext_vector_type(8)))  __bf16 v8bf;
typedef __attribute__((ext_vector_type(4)))  __bf16 v4bf;
typedef __attribute__((ext_vector_type(8)))  float  v8f;

static __device__ __forceinline__ v8f wmma_bf16(v16bf a, v16bf b, v8f c) {
  // D = A(16x32 bf16) x B(32x16 bf16) + C(16x16 f32)
  return __builtin_amdgcn_wmma_f32_16x16x32_bf16(false, a, false, b, (short)0, c, false, false);
}

// ---------------------------------------------------------------------------
// Generic tiled WMMA GEMM: C = epilogue(A[MxK] @ B[KxN])
// epilogue: v = (acc + bias[col]) * (alpha * gate[0]) + res[row,col]
// A,B row-major; templated fp32/bf16 operand reads (converted to bf16 in LDS).
// BM=128, BN=128, BK=32, 256 threads = 8 waves laid out 4 (M) x 2 (N);
// each wave computes 32x64 = 2x4 WMMA tiles.
// Double-buffered LDS: global loads for stage k+1 are issued before the WMMAs
// of stage k; global_prefetch covers stage k+2.
// ---------------------------------------------------------------------------
#define BM 128
#define BN 128
#define BKK 32
#define LPAD 8

template <bool AF32, bool BF32>
__global__ __launch_bounds__(256) void gemm_kernel(
    const void* __restrict__ Ap, const void* __restrict__ Bp, int M, int N,
    int K, const float* __restrict__ bias, const float* res,
    const float* __restrict__ gate, float alpha, float* Cf, bf16* Cb) {
  __shared__ bf16 As[2][BM][BKK + LPAD];
  __shared__ bf16 Bs[2][BN][BKK + LPAD];  // stored transposed: Bs[.][n][k]

  const int t = threadIdx.x;
  const int w = t >> 5;
  const int lane = t & 31;
  const int half = lane >> 4;
  const int lr = lane & 15;
  const int wm = (w & 3) * 32;
  const int wn = (w >> 2) * 64;
  const long bm = (long)blockIdx.y * BM;
  const long bn = (long)blockIdx.x * BN;

  const int arow = (t >> 3);        // + it*32
  const int acol = (t & 7) * 4;
  const int brow = (t >> 5);        // + it*8 (k index)
  const int bcol = (t & 31) * 4;

  v8f acc[2][4];
#pragma unroll
  for (int i = 0; i < 2; ++i)
#pragma unroll
    for (int j = 0; j < 4; ++j)
#pragma unroll
      for (int e = 0; e < 8; ++e) acc[i][j][e] = 0.0f;

  // register staging for the in-flight tile
  float4 avf[4], bvf[4];
  v4bf avb[4], bvb[4];

  auto load_tiles = [&](int k0) {
#pragma unroll
    for (int it = 0; it < 4; ++it) {
      if (AF32)
        avf[it] = *(const float4*)((const float*)Ap +
                                   (bm + arow + it * 32) * (long)K + k0 + acol);
      else
        avb[it] = *(const v4bf*)((const bf16*)Ap +
                                 (bm + arow + it * 32) * (long)K + k0 + acol);
      if (BF32)
        bvf[it] = *(const float4*)((const float*)Bp +
                                   (long)(k0 + brow + it * 8) * N + bn + bcol);
      else
        bvb[it] = *(const v4bf*)((const bf16*)Bp +
                                 (long)(k0 + brow + it * 8) * N + bn + bcol);
    }
  };
  auto store_tiles = [&](int buf) {
#pragma unroll
    for (int it = 0; it < 4; ++it) {
      v4bf a4;
      if (AF32) {
        a4[0] = (bf16)avf[it].x; a4[1] = (bf16)avf[it].y;
        a4[2] = (bf16)avf[it].z; a4[3] = (bf16)avf[it].w;
      } else {
        a4 = avb[it];
      }
      *(v4bf*)&As[buf][arow + it * 32][acol] = a4;  // one ds_store_b64
      bf16 b4[4];
      if (BF32) {
        b4[0] = (bf16)bvf[it].x; b4[1] = (bf16)bvf[it].y;
        b4[2] = (bf16)bvf[it].z; b4[3] = (bf16)bvf[it].w;
      } else {
        b4[0] = bvb[it][0]; b4[1] = bvb[it][1];
        b4[2] = bvb[it][2]; b4[3] = bvb[it][3];
      }
      // transposed scatter (unavoidable scalar stores)
      Bs[buf][bcol + 0][brow + it * 8] = b4[0];
      Bs[buf][bcol + 1][brow + it * 8] = b4[1];
      Bs[buf][bcol + 2][brow + it * 8] = b4[2];
      Bs[buf][bcol + 3][brow + it * 8] = b4[3];
    }
  };

  const int ksteps = K / BKK;
  load_tiles(0);
  store_tiles(0);
  __syncthreads();

  for (int ks = 0; ks < ksteps; ++ks) {
    const int buf = ks & 1;
    // issue next stage's global loads before computing this stage
    if (ks + 1 < ksteps) load_tiles((ks + 1) * BKK);
    // prefetch stage k+2 into caches (global_prefetch_b8)
    if (ks + 2 < ksteps) {
      const int k2 = (ks + 2) * BKK;
      if (AF32)
        __builtin_prefetch((const float*)Ap + (bm + arow) * (long)K + k2 + acol);
      else
        __builtin_prefetch((const bf16*)Ap + (bm + arow) * (long)K + k2 + acol);
      if (BF32)
        __builtin_prefetch((const float*)Bp + (long)(k2 + brow) * N + bn + bcol);
      else
        __builtin_prefetch((const bf16*)Bp + (long)(k2 + brow) * N + bn + bcol);
    }

    // ---- fragments per ISA layouts ----
    v16bf afr[2], bfr[4];
#pragma unroll
    for (int i = 0; i < 2; ++i) {
      // A 16x32: lane lr = row; halves {0..7}->K kb..kb+7, {8..15}->K kb+16..
      const bf16* p = &As[buf][wm + i * 16 + lr][half * 8];
      v8bf lo = *(const v8bf*)p;
      v8bf hi = *(const v8bf*)(p + 16);
#pragma unroll
      for (int j = 0; j < 8; ++j) { afr[i][j] = lo[j]; afr[i][8 + j] = hi[j]; }
    }
#pragma unroll
    for (int j4 = 0; j4 < 4; ++j4) {
      // B 32x16: lane lr = col; halves {0..7}->K half*8.., {8..15}->K 16+half*8..
      const bf16* p = &Bs[buf][wn + j4 * 16 + lr][half * 8];
      v8bf lo = *(const v8bf*)p;
      v8bf hi = *(const v8bf*)(p + 16);
#pragma unroll
      for (int j = 0; j < 8; ++j) { bfr[j4][j] = lo[j]; bfr[j4][8 + j] = hi[j]; }
    }
#pragma unroll
    for (int i = 0; i < 2; ++i)
#pragma unroll
      for (int j = 0; j < 4; ++j)
        acc[i][j] = wmma_bf16(afr[i], bfr[j], acc[i][j]);

    // stage k+1 into the other buffer (different memory than being read)
    if (ks + 1 < ksteps) store_tiles(buf ^ 1);
    __syncthreads();
  }

  const float gsc = alpha * (gate ? gate[0] : 1.0f);
#pragma unroll
  for (int i = 0; i < 2; ++i) {
#pragma unroll
    for (int j = 0; j < 4; ++j) {
      long col = bn + wn + j * 16 + lr;
      float bv = bias ? bias[col] : 0.0f;
#pragma unroll
      for (int r = 0; r < 8; ++r) {
        long row = bm + wm + i * 16 + half * 8 + r;
        long idx = row * (long)N + col;
        float v = (acc[i][j][r] + bv) * gsc;
        if (res) v += res[idx];
        if (Cf) Cf[idx] = v;
        if (Cb) Cb[idx] = (bf16)v;
      }
    }
  }
}

// ---------------------------------------------------------------------------
// Fused flash-style attention: one block = (batch b, head h, 16 query rows).
// q: (B,256,512) bf16 (dh^-0.5 pre-folded); kv: (B,MC,1024) bf16 (k|v).
// MC is a template constant so every accumulator loop fully unrolls and the
// v8f accumulator arrays stay in VGPRs.
// ---------------------------------------------------------------------------
template <int MC>
__global__ __launch_bounds__(256) void attn_kernel(const bf16* __restrict__ q,
                                                   const bf16* __restrict__ kv,
                                                   bf16* __restrict__ o) {
  constexpr int MCW = MC >> 3;   // columns per wave (128 or 32)
  constexpr int NT  = MCW >> 4;  // 16-wide score tiles per wave (8 or 2)
  constexpr int KS3 = MCW >> 5;  // 32-wide P@V k-steps per wave (4 or 1)

  __shared__ bf16 P[16][MC + 8];
  __shared__ bf16 Op[8][16][64];
  __shared__ float red[8][16];

  const int t = threadIdx.x, w = t >> 5, lane = t & 31;
  const int half = lane >> 4, lr = lane & 15;
  const int b = blockIdx.y >> 3, h = blockIdx.y & 7;
  const int q0 = blockIdx.x * 16;
  const int colbase = w * MCW;

  // q A-fragments, K=64 -> two 32-wide k-steps
  const bf16* qrow = q + ((long)(b * 256 + q0 + lr) * 512) + h * 64;
  v16bf a0, a1;
  {
    v8bf l0 = *(const v8bf*)(qrow + half * 8);
    v8bf h0 = *(const v8bf*)(qrow + half * 8 + 16);
    v8bf l1 = *(const v8bf*)(qrow + 32 + half * 8);
    v8bf h1 = *(const v8bf*)(qrow + 32 + half * 8 + 16);
#pragma unroll
    for (int j = 0; j < 8; ++j) {
      a0[j] = l0[j]; a0[8 + j] = h0[j];
      a1[j] = l1[j]; a1[8 + j] = h1[j];
    }
  }

  v8f acc[NT];
#pragma unroll
  for (int i = 0; i < NT; ++i)
#pragma unroll
    for (int e = 0; e < 8; ++e) acc[i][e] = 0.0f;

  // ---- Phase 1: scores = q @ k^T ----
#pragma unroll
  for (int tt = 0; tt < NT; ++tt) {
    int ctx = colbase + tt * 16 + lr;
    const bf16* kr = kv + ((long)(b * MC + ctx) * 1024) + h * 64;
    v16bf b0, b1;
    v8bf x0 = *(const v8bf*)(kr + half * 8);
    v8bf x1 = *(const v8bf*)(kr + half * 8 + 16);
    v8bf y0 = *(const v8bf*)(kr + 32 + half * 8);
    v8bf y1 = *(const v8bf*)(kr + 32 + half * 8 + 16);
#pragma unroll
    for (int j = 0; j < 8; ++j) {
      b0[j] = x0[j]; b0[8 + j] = x1[j];
      b1[j] = y0[j]; b1[8 + j] = y1[j];
    }
    acc[tt] = wmma_bf16(a0, b0, acc[tt]);
    acc[tt] = wmma_bf16(a1, b1, acc[tt]);
  }

  // ---- Phase 2: softmax over MC ----
  float rowm[8];
#pragma unroll
  for (int r = 0; r < 8; ++r) rowm[r] = -1e30f;
#pragma unroll
  for (int tt = 0; tt < NT; ++tt)
#pragma unroll
    for (int r = 0; r < 8; ++r) rowm[r] = fmaxf(rowm[r], acc[tt][r]);
#pragma unroll
  for (int off = 1; off < 16; off <<= 1)
#pragma unroll
    for (int r = 0; r < 8; ++r)
      rowm[r] = fmaxf(rowm[r], __shfl_xor(rowm[r], off, 32));
  if (lr == 0)
#pragma unroll
    for (int r = 0; r < 8; ++r) red[w][half * 8 + r] = rowm[r];
  __syncthreads();
#pragma unroll
  for (int r = 0; r < 8; ++r) {
    float mx = red[0][half * 8 + r];
#pragma unroll
    for (int ww = 1; ww < 8; ++ww) mx = fmaxf(mx, red[ww][half * 8 + r]);
    rowm[r] = mx;
  }
  __syncthreads();
  float rows[8];
#pragma unroll
  for (int r = 0; r < 8; ++r) rows[r] = 0.0f;
#pragma unroll
  for (int tt = 0; tt < NT; ++tt)
#pragma unroll
    for (int r = 0; r < 8; ++r) {
      float e = __expf(acc[tt][r] - rowm[r]);
      acc[tt][r] = e;
      rows[r] += e;
    }
#pragma unroll
  for (int off = 1; off < 16; off <<= 1)
#pragma unroll
    for (int r = 0; r < 8; ++r) rows[r] += __shfl_xor(rows[r], off, 32);
  if (lr == 0)
#pragma unroll
    for (int r = 0; r < 8; ++r) red[w][half * 8 + r] = rows[r];
  __syncthreads();
#pragma unroll
  for (int r = 0; r < 8; ++r) {
    float s = 0.0f;
#pragma unroll
    for (int ww = 0; ww < 8; ++ww) s += red[ww][half * 8 + r];
    rows[r] = 1.0f / s;
  }
#pragma unroll
  for (int tt = 0; tt < NT; ++tt)
#pragma unroll
    for (int r = 0; r < 8; ++r)
      P[half * 8 + r][colbase + tt * 16 + lr] = (bf16)(acc[tt][r] * rows[r]);
  __syncthreads();

  // ---- Phase 3: o = P @ v (each wave owns a k-chunk, partial 16x64) ----
  v8f oacc[4];
#pragma unroll
  for (int i = 0; i < 4; ++i)
#pragma unroll
    for (int e = 0; e < 8; ++e) oacc[i][e] = 0.0f;
#pragma unroll
  for (int s = 0; s < KS3; ++s) {
    int kg = colbase + s * 32;
    const bf16* pr = &P[lr][kg + half * 16];
    v8bf lo = *(const v8bf*)pr;
    v8bf hi = *(const v8bf*)(pr + 8);
    v16bf af;
#pragma unroll
    for (int j = 0; j < 8; ++j) { af[j] = lo[j]; af[8 + j] = hi[j]; }
#pragma unroll
    for (int n4 = 0; n4 < 4; ++n4) {
      const bf16* vb = kv + ((long)(b * MC + kg + half * 16)) * 1024 + 512 +
                       h * 64 + n4 * 16 + lr;
      v16bf bf_;
#pragma unroll
      for (int j = 0; j < 16; ++j) bf_[j] = vb[(long)j * 1024];
      oacc[n4] = wmma_bf16(af, bf_, oacc[n4]);
    }
  }
#pragma unroll
  for (int n4 = 0; n4 < 4; ++n4)
#pragma unroll
    for (int r = 0; r < 8; ++r)
      Op[w][half * 8 + r][n4 * 16 + lr] = (bf16)oacc[n4][r];
  __syncthreads();
  for (int idx = t; idx < 1024; idx += 256) {
    int m = idx >> 6, n = idx & 63;
    float s = 0.0f;
#pragma unroll
    for (int ww = 0; ww < 8; ++ww) s += (float)Op[ww][m][n];
    o[((long)(b * 256 + q0 + m) * 512) + h * 64 + n] = (bf16)s;
  }
}

// ---------------------------------------------------------------------------
// LayerNorm helpers (row length 512, 256 threads/row -> 2 elems/thread)
// ---------------------------------------------------------------------------
static __device__ __forceinline__ float block_sum(float v, float* sm, int t) {
  sm[t] = v;
  __syncthreads();
  for (int off = 128; off; off >>= 1) {
    if (t < off) sm[t] += sm[t + off];
    __syncthreads();
  }
  float r = sm[0];
  __syncthreads();
  return r;
}

__global__ __launch_bounds__(256) void ln_kernel(const float* __restrict__ X,
                                                 const float* __restrict__ g,
                                                 const float* __restrict__ bb,
                                                 bf16* __restrict__ out) {
  __shared__ float sm[256];
  const int row = blockIdx.x, t = threadIdx.x;
  const float* xr = X + (long)row * 512;
  float v0 = xr[t], v1 = xr[t + 256];
  float mu = block_sum(v0 + v1, sm, t) * (1.0f / 512.0f);
  float d0 = v0 - mu, d1 = v1 - mu;
  float var = block_sum(d0 * d0 + d1 * d1, sm, t) * (1.0f / 512.0f);
  float rsq = rsqrtf(var + 1e-5f);
  out[(long)row * 512 + t] = (bf16)(d0 * rsq * g[t] + bb[t]);
  out[(long)row * 512 + 256 + t] = (bf16)(d1 * rsq * g[t + 256] + bb[t + 256]);
}

// ctx = LN(y; g,b) then normalize(ctx) (param-free) -> bf16
__global__ __launch_bounds__(256) void double_ln_kernel(
    const float* __restrict__ X, const float* __restrict__ g,
    const float* __restrict__ bb, bf16* __restrict__ out) {
  __shared__ float sm[256];
  const int row = blockIdx.x, t = threadIdx.x;
  const float* xr = X + (long)row * 512;
  float v0 = xr[t], v1 = xr[t + 256];
  float mu = block_sum(v0 + v1, sm, t) * (1.0f / 512.0f);
  float d0 = v0 - mu, d1 = v1 - mu;
  float var = block_sum(d0 * d0 + d1 * d1, sm, t) * (1.0f / 512.0f);
  float rsq = rsqrtf(var + 1e-5f);
  float y0 = d0 * rsq * g[t] + bb[t];
  float y1 = d1 * rsq * g[t + 256] + bb[t + 256];
  float mu2 = block_sum(y0 + y1, sm, t) * (1.0f / 512.0f);
  float e0 = y0 - mu2, e1 = y1 - mu2;
  float var2 = block_sum(e0 * e0 + e1 * e1, sm, t) * (1.0f / 512.0f);
  float rsq2 = rsqrtf(var2 + 1e-5f);
  out[(long)row * 512 + t] = (bf16)(e0 * rsq2);
  out[(long)row * 512 + 256 + t] = (bf16)(e1 * rsq2);
}

// Fold per-layer ctx-LN affine into wkv:  W'[k,j]=g[k]*wkv[k,j],  bias'[j]=sum_k b[k]*wkv[k,j]
__global__ __launch_bounds__(256) void fold_kv_kernel(
    const float* __restrict__ g, const float* __restrict__ bb,
    const float* __restrict__ wkv, bf16* __restrict__ wp,
    float* __restrict__ biasp) {
  const int j = blockIdx.x * 256 + threadIdx.x;  // 0..1023
  float acc = 0.0f;
  for (int k = 0; k < 512; ++k) {
    float wv = wkv[(long)k * 1024 + j];
    acc += bb[k] * wv;
    wp[(long)k * 1024 + j] = (bf16)(g[k] * wv);
  }
  biasp[j] = acc;
}

// act = a * selu(gate) from h (rows x 1024, split 512|512) -> bf16 (rows x 512)
__global__ __launch_bounds__(256) void gated_selu_kernel(
    const float* __restrict__ h, bf16* __restrict__ act) {
  const int idx = blockIdx.x * 256 + threadIdx.x;
  const int row = idx >> 9, col = idx & 511;
  float a = h[(long)row * 1024 + col];
  float gg = h[(long)row * 1024 + 512 + col];
  const float sc = 1.0507009873554805f, al = 1.6732632423543772f;
  float s = gg > 0.0f ? sc * gg : sc * al * (__expf(gg) - 1.0f);
  act[idx] = (bf16)(a * s);
}

__global__ __launch_bounds__(256) void init_x_kernel(
    const float* __restrict__ latents, float* __restrict__ x) {
  const int idx = blockIdx.x * 256 + threadIdx.x;  // 16*256*512
  x[idx] = latents[idx & 131071];
}

__global__ __launch_bounds__(256) void mean_kernel(const float* __restrict__ x,
                                                   float* __restrict__ out) {
  const int idx = blockIdx.x * 256 + threadIdx.x;  // 16*512
  const int b = idx >> 9, d = idx & 511;
  const float* p = x + (long)b * 131072 + d;
  float s = 0.0f;
  for (int n = 0; n < 256; ++n) s += p[(long)n * 512];
  out[idx] = s * (1.0f / 256.0f);
}

// ---------------------------------------------------------------------------
extern "C" void kernel_launch(void* const* d_in, const int* in_sizes, int n_in,
                              void* d_out, int out_size, void* d_ws,
                              size_t ws_size, hipStream_t stream) {
  (void)in_sizes; (void)n_in; (void)out_size; (void)ws_size;
  const float* emb0      = (const float*)d_in[0];
  const float* emb1      = (const float*)d_in[1];
  const float* latents   = (const float*)d_in[2];
  const float* proj_w    = (const float*)d_in[3];
  const float* proj_b    = (const float*)d_in[4];
  const float* proj_g    = (const float*)d_in[5];
  const float* proj_beta = (const float*)d_in[6];
  const float* ca_ln_g   = (const float*)d_in[7];
  const float* ca_ln_b   = (const float*)d_in[8];
  const float* ca_ctx_g  = (const float*)d_in[9];
  const float* ca_ctx_b  = (const float*)d_in[10];
  const float* ca_wq     = (const float*)d_in[11];
  const float* ca_wkv    = (const float*)d_in[12];
  const float* ca_wo     = (const float*)d_in[13];
  const float* ca_bo     = (const float*)d_in[14];
  const float* ca_gate   = (const float*)d_in[15];
  const float* cf_ln_g   = (const float*)d_in[16];
  const float* cf_ln_b   = (const float*)d_in[17];
  const float* cf_w1     = (const float*)d_in[18];
  const float* cf_b1     = (const float*)d_in[19];
  const float* cf_w2     = (const float*)d_in[20];
  const float* cf_b2     = (const float*)d_in[21];
  const float* la_ln_g   = (const float*)d_in[22];
  const float* la_ln_b   = (const float*)d_in[23];
  const float* la_wq     = (const float*)d_in[24];
  const float* la_wkv    = (const float*)d_in[25];
  const float* la_wo     = (const float*)d_in[26];
  const float* la_bo     = (const float*)d_in[27];
  const float* la_gate   = (const float*)d_in[28];
  const float* lf_ln_g   = (const float*)d_in[29];
  const float* lf_ln_b   = (const float*)d_in[30];
  const float* lf_w1     = (const float*)d_in[31];
  const float* lf_b1     = (const float*)d_in[32];
  const float* lf_w2     = (const float*)d_in[33];
  const float* lf_b2     = (const float*)d_in[34];

  char* ws = (char*)d_ws;
  size_t off = 0;
  auto alloc = [&](size_t bytes) -> void* {
    void* p = ws + off;
    off += (bytes + 255) & ~(size_t)255;
    return p;
  };
  float* ctx_tmp = (float*)alloc(16384ull * 512 * 4);
  bf16*  nctx0   = (bf16*)alloc(16384ull * 512 * 2);
  bf16*  nctx1   = (bf16*)alloc(16384ull * 512 * 2);
  float* x       = (float*)alloc(4096ull * 512 * 4);
  bf16*  xn      = (bf16*)alloc(4096ull * 512 * 2);
  bf16*  qb      = (bf16*)alloc(4096ull * 512 * 2);
  bf16*  kvb     = (bf16*)alloc(16384ull * 1024 * 2);
  bf16*  ob      = (bf16*)alloc(4096ull * 512 * 2);
  float* hbuf    = (float*)alloc(4096ull * 1024 * 4);
  bf16*  act     = (bf16*)alloc(4096ull * 512 * 2);
  bf16*  wkvp    = (bf16*)alloc(512ull * 1024 * 2);
  float* biasp   = (float*)alloc(1024ull * 4);

  bf16* nctx[2] = {nctx0, nctx1};
  const float* emb[2] = {emb0, emb1};

  // Stage 0: ctx_m = normalize(LN(emb_m @ proj_w + proj_b; proj_g, proj_beta))
  for (int m = 0; m < 2; ++m) {
    gemm_kernel<true, true><<<dim3(4, 128), 256, 0, stream>>>(
        emb[m], proj_w, 16384, 512, 1024, proj_b, nullptr, nullptr, 1.0f,
        ctx_tmp, nullptr);
    double_ln_kernel<<<16384, 256, 0, stream>>>(ctx_tmp, proj_g, proj_beta,
                                                nctx[m]);
  }
  init_x_kernel<<<8192, 256, 0, stream>>>(latents, x);

  for (int i = 0; i < 4; ++i) {
    for (int m = 0; m < 2; ++m) {
      const int lm = i * 2 + m;
      // cross-attention
      ln_kernel<<<4096, 256, 0, stream>>>(x, ca_ln_g + lm * 512,
                                          ca_ln_b + lm * 512, xn);
      gemm_kernel<false, true><<<dim3(4, 32), 256, 0, stream>>>(
          xn, ca_wq + (size_t)lm * 512 * 512, 4096, 512, 512, nullptr, nullptr,
          nullptr, 0.125f, nullptr, qb);
      fold_kv_kernel<<<4, 256, 0, stream>>>(ca_ctx_g + lm * 512,
                                            ca_ctx_b + lm * 512,
                                            ca_wkv + (size_t)lm * 512 * 1024,
                                            wkvp, biasp);
      gemm_kernel<false, false><<<dim3(8, 128), 256, 0, stream>>>(
          nctx[m], wkvp, 16384, 1024, 512, biasp, nullptr, nullptr, 1.0f,
          nullptr, kvb);
      attn_kernel<1024><<<dim3(16, 128), 256, 0, stream>>>(qb, kvb, ob);
      gemm_kernel<false, true><<<dim3(4, 32), 256, 0, stream>>>(
          ob, ca_wo + (size_t)lm * 512 * 512, 4096, 512, 512, ca_bo + lm * 512,
          x, ca_gate + lm, 1.0f, x, nullptr);
      // FFN
      ln_kernel<<<4096, 256, 0, stream>>>(x, cf_ln_g + lm * 512,
                                          cf_ln_b + lm * 512, xn);
      gemm_kernel<false, true><<<dim3(8, 32), 256, 0, stream>>>(
          xn, cf_w1 + (size_t)lm * 512 * 1024, 4096, 1024, 512,
          cf_b1 + lm * 1024, nullptr, nullptr, 1.0f, hbuf, nullptr);
      gated_selu_kernel<<<8192, 256, 0, stream>>>(hbuf, act);
      gemm_kernel<false, true><<<dim3(4, 32), 256, 0, stream>>>(
          act, cf_w2 + (size_t)lm * 512 * 512, 4096, 512, 512, cf_b2 + lm * 512,
          x, nullptr, 1.0f, x, nullptr);
    }
    // latent self-attention
    ln_kernel<<<4096, 256, 0, stream>>>(x, la_ln_g + i * 512,
                                        la_ln_b + i * 512, xn);
    gemm_kernel<false, true><<<dim3(4, 32), 256, 0, stream>>>(
        xn, la_wq + (size_t)i * 512 * 512, 4096, 512, 512, nullptr, nullptr,
        nullptr, 0.125f, nullptr, qb);
    gemm_kernel<false, true><<<dim3(8, 32), 256, 0, stream>>>(
        xn, la_wkv + (size_t)i * 512 * 1024, 4096, 1024, 512, nullptr, nullptr,
        nullptr, 1.0f, nullptr, kvb);
    attn_kernel<256><<<dim3(16, 128), 256, 0, stream>>>(qb, kvb, ob);
    gemm_kernel<false, true><<<dim3(4, 32), 256, 0, stream>>>(
        ob, la_wo + (size_t)i * 512 * 512, 4096, 512, 512, la_bo + i * 512, x,
        la_gate + i, 1.0f, x, nullptr);
    // latent FFN
    ln_kernel<<<4096, 256, 0, stream>>>(x, lf_ln_g + i * 512,
                                        lf_ln_b + i * 512, xn);
    gemm_kernel<false, true><<<dim3(8, 32), 256, 0, stream>>>(
        xn, lf_w1 + (size_t)i * 512 * 1024, 4096, 1024, 512, lf_b1 + i * 1024,
        nullptr, nullptr, 1.0f, hbuf, nullptr);
    gated_selu_kernel<<<8192, 256, 0, stream>>>(hbuf, act);
    gemm_kernel<false, true><<<dim3(4, 32), 256, 0, stream>>>(
        act, lf_w2 + (size_t)i * 512 * 512, 4096, 512, 512, lf_b2 + i * 512, x,
        nullptr, 1.0f, x, nullptr);
  }
  mean_kernel<<<32, 256, 0, stream>>>(x, (float*)d_out);
}
